// AddSymbolsEncodingsToExpressions_32134945308867
// MI455X (gfx1250) — matrix operand order repaired and checked by reading, not compiled, not run on hardware
//
#include <hip/hip_runtime.h>
#include <hip/hip_bf16.h>

typedef __attribute__((ext_vector_type(16))) __bf16 v16bf;
typedef __attribute__((ext_vector_type(8)))  float  v8f;

union FragB { uint4 q[2]; v16bf v; };

__device__ __forceinline__ unsigned int pack_bf16x2(float a, float b) {
  unsigned int ua = __float_as_uint(a);
  unsigned int ub = __float_as_uint(b);
  ua = (ua + 0x7FFFu + ((ua >> 16) & 1u)) >> 16;  // round-to-nearest-even
  ub = (ub + 0x7FFFu + ((ub >> 16) & 1u)) >> 16;
  return ua | (ub << 16);
}

// ---------------------------------------------------------------------------
// Prepack Wf/Wa (f32 [256 x 512]) into bf16 WMMA B-fragments in d_ws.
// Fragment (ntile, kk) = 32 lanes x 32 bytes (1 KB). Lane l, halfword h maps
// to B[k, n] = W[n, k] with n = ntile*16 + (l&15), k = kk*32 + 16*(l>>4) + h.
// ---------------------------------------------------------------------------
__global__ void prepack_weights(const float* __restrict__ Wf,
                                const float* __restrict__ Wa,
                                unsigned int* __restrict__ ws) {
  int tid   = blockIdx.x * blockDim.x + threadIdx.x;   // 16384 total
  int w     = tid >> 13;
  int rem   = tid & 8191;
  int ntile = rem >> 9;
  int kk    = (rem >> 5) & 15;
  int lane  = rem & 31;
  const float* W = w ? Wa : Wf;
  int n  = ntile * 16 + (lane & 15);
  int kb = kk * 32 + 16 * (lane >> 4);
  unsigned int o[8];
#pragma unroll
  for (int p = 0; p < 8; ++p) {
    float a = W[n * 512 + kb + 2 * p];
    float b = W[n * 512 + kb + 2 * p + 1];
    o[p] = pack_bf16x2(a, b);
  }
  unsigned int* dst = ws + w * 65536 + ((ntile * 16 + kk) * 256 + lane * 8);
  *(uint4*)(dst)     = make_uint4(o[0], o[1], o[2], o[3]);
  *(uint4*)(dst + 4) = make_uint4(o[4], o[5], o[6], o[7]);
}

// ---------------------------------------------------------------------------
// Main kernel: 32 threads = 1 wave per block; wave handles M=32 rows as two
// 16-row A tiles that share every B fragment (4 WMMAs per B-pair load),
// halving weight re-fetch traffic vs M=16. 32 KB LDS A-stage per block.
// ---------------------------------------------------------------------------
__global__ void __launch_bounds__(32)
gated_update(const float* __restrict__ flat,      // [262144, 256]
             const float* __restrict__ symbols,   // [20000, 256]
             const float* __restrict__ bfv,       // [256]
             const float* __restrict__ bav,       // [256]
             const int*   __restrict__ expr_idx,  // [N_OCC]
             const int*   __restrict__ token_idx, // [N_OCC]
             const int*   __restrict__ symbol_idx,// [N_OCC]
             const unsigned int* __restrict__ wsW,// packed weights (2 gates)
             float* __restrict__ out) {
  __shared__ unsigned int ldsA[2][4096];   // 2 row-tiles x 16 kk-frags x 1KB

  const int lane    = threadIdx.x;          // 0..31
  const int rowBase = blockIdx.x * 32;

  // each lane owns one row's scatter indices
  const int myRow = rowBase + lane;
  const int occ   = expr_idx[myRow] * 64 + token_idx[myRow];
  const int sym   = symbol_idx[myRow];

  // ---- gather prev|upd rows, convert f32->bf16, stage in A-fragment layout
  for (int t = 0; t < 2; ++t) {
    for (int m = 0; m < 16; ++m) {
      int srcOcc = __shfl(occ, t * 16 + m, 32);   // ds_bpermute broadcast
      int srcSym = __shfl(sym, t * 16 + m, 32);
#pragma unroll
      for (int c = 0; c < 8; ++c) {
        int k = c * 64 + lane * 2;                 // concat column (0..511)
        const float* src = (c < 4)
            ? (flat    + (long)srcOcc * 256 + k)
            : (symbols + (long)srcSym * 256 + (k - 256));
        float2 v = *(const float2*)src;            // coalesced b64 per lane
        int ko = k & 31;
        int kk = k >> 5;
        int laneDst = m + 16 * (ko >> 4);          // contiguous k: k = 16*half + h
        int h       = ko & 15;
        ldsA[t][kk * 256 + laneDst * 8 + (h >> 1)] = pack_bf16x2(v.x, v.y);
      }
    }
  }
  __syncthreads();   // single-wave WG: barrier is free; DS in-order per wave

  const int colLo = lane & 15;
  const int hi    = lane >> 4;

  // precompute 32-bit scatter offsets for the epilogue (C/D row M=j+8*hi)
  int offP[2][8], offS[2][8];
#pragma unroll
  for (int t = 0; t < 2; ++t)
#pragma unroll
    for (int j = 0; j < 8; ++j) {
      int mrow = t * 16 + j + 8 * hi;
      offP[t][j] = __shfl(occ, mrow, 32) * 256;
      offS[t][j] = __shfl(sym, mrow, 32) * 256;
    }

  for (int n = 0; n < 16; ++n) {                   // 16 output col tiles
    v8f accF[2] = {{}, {}};
    v8f accA[2] = {{}, {}};
#pragma unroll 2
    for (int kk = 0; kk < 16; ++kk) {              // K = 512 in steps of 32
      FragB a0, a1, bF, bA;
      const uint4* ap0 = (const uint4*)&ldsA[0][kk * 256 + lane * 8];
      const uint4* ap1 = (const uint4*)&ldsA[1][kk * 256 + lane * 8];
      a0.q[0] = ap0[0]; a0.q[1] = ap0[1];          // ds_load_b128 x2
      a1.q[0] = ap1[0]; a1.q[1] = ap1[1];
      const unsigned int* woff = wsW + ((n * 16 + kk) * 256 + lane * 8);
      const uint4* fp = (const uint4*)(woff);
      const uint4* gp = (const uint4*)(woff + 65536);
      bF.q[0] = fp[0]; bF.q[1] = fp[1];            // global_load_b128 x2
      bA.q[0] = gp[0]; bA.q[1] = gp[1];
      // 4 WMMAs reuse each B-fragment pair across both row tiles
      accF[0] = __builtin_amdgcn_wmma_f32_16x16x32_bf16(
          false, a0.v, false, bF.v, (short)0, accF[0], false, false);
      accA[0] = __builtin_amdgcn_wmma_f32_16x16x32_bf16(
          false, a0.v, false, bA.v, (short)0, accA[0], false, false);
      accF[1] = __builtin_amdgcn_wmma_f32_16x16x32_bf16(
          false, a1.v, false, bF.v, (short)0, accF[1], false, false);
      accA[1] = __builtin_amdgcn_wmma_f32_16x16x32_bf16(
          false, a1.v, false, bA.v, (short)0, accA[1], false, false);
    }
    int col = n * 16 + colLo;
    float biasF = bfv[col];
    float biasA = bav[col];
#pragma unroll
    for (int t = 0; t < 2; ++t) {
#pragma unroll
      for (int j = 0; j < 8; ++j) {                // C/D layout: M=j+8*hi
        float prevv = flat[offP[t][j] + col];      // L2-hot re-gather (f32)
        float updv  = symbols[offS[t][j] + col];
        float xf = accF[t][j] + biasF;
        float xa = accA[t][j] + biasA;
        float sf = 1.0f / (1.0f + __expf(-xf));
        float sa = 1.0f / (1.0f + __expf(-xa));
        out[offP[t][j] + col] = sf * prevv + sa * updv;  // unique rows
      }
    }
  }
}

extern "C" void kernel_launch(void* const* d_in, const int* in_sizes, int n_in,
                              void* d_out, int out_size, void* d_ws, size_t ws_size,
                              hipStream_t stream) {
  const float* expr = (const float*)d_in[0];
  const float* syms = (const float*)d_in[1];
  const float* Wf   = (const float*)d_in[2];
  const float* bf   = (const float*)d_in[3];
  const float* Wa   = (const float*)d_in[4];
  const float* ba   = (const float*)d_in[5];
  const int*   eidx = (const int*)d_in[6];
  const int*   tidx = (const int*)d_in[7];
  const int*   sidx = (const int*)d_in[8];
  float* out        = (float*)d_out;
  unsigned int* ws  = (unsigned int*)d_ws;

  // out = copy of expressions (scatter base), ordered before the update kernel
  hipMemcpyAsync(out, expr, (size_t)4096 * 64 * 256 * sizeof(float),
                 hipMemcpyDeviceToDevice, stream);

  // bf16 weight prepack into WMMA fragment order (2 x 256 KB in d_ws)
  prepack_weights<<<64, 256, 0, stream>>>(Wf, Wa, ws);

  // 131072 rows / 32 rows per wave-block
  gated_update<<<131072 / 32, 32, 0, stream>>>(expr, syms, bf, ba,
                                               eidx, tidx, sidx, ws, out);
}
